// KAF_Layer_69054484185847
// MI455X (gfx1250) — matrix operand order repaired and checked by reading, not compiled, not run on hardware
//
#include <hip/hip_runtime.h>
#include <hip/hip_bf16.h>
#include <math.h>

typedef _Float16 F16;
typedef __attribute__((ext_vector_type(16))) _Float16 v16h;
typedef __attribute__((ext_vector_type(8)))  float    v8f;

// ---------------------------------------------------------------------------
// WMMA tile helper: one wave computes a 16(rows) x 64 output tile.
//   feat16 : LDS, 16 rows of this wave, row-major, row stride = KP halfs
//   wcm    : LDS, weights column-major [64][KP]
//   acc[4] : f32 accumulators, acc[nt] covers cols nt*16..nt*16+15
// A/B k order here: k = kk + 16*(lane/16) + i in both (16-bit A 16x32, B 32x16, f32 C/D).
// ---------------------------------------------------------------------------
template<int KP>
__device__ inline void mma_tile(const F16* feat16, const F16* wcm, v8f* acc, int lane) {
  const int hi = lane >> 4, lo = lane & 15;
  const F16* rp = feat16 + lo * KP;
  for (int kk = 0; kk < KP; kk += 32) {
    v16h a;
    const int ka = kk + hi * 16;
#pragma unroll
    for (int i = 0; i < 16; i++) a[i] = rp[ka + i];
#pragma unroll
    for (int nt = 0; nt < 4; nt++) {
      const F16* cp = wcm + (size_t)(nt * 16 + lo) * KP;
      v16h b;
      const int kb = kk + hi * 16;
#pragma unroll
      for (int i = 0; i < 16; i++) b[i] = cp[kb + i];
      acc[nt] = __builtin_amdgcn_wmma_f32_16x16x32_f16(
          false, a, false, b, (short)0, acc[nt], false, false);
    }
  }
}

// +bias, SiLU, store f16 to mid tile (16 rows of this wave, row stride 64)
__device__ inline void epilogue_silu(v8f* acc, const float* b, F16* mid, int lane) {
  const int hi = lane >> 4, lo = lane & 15;
#pragma unroll
  for (int nt = 0; nt < 4; nt++) {
    const int col = nt * 16 + lo;
    const float bb = b[col];
#pragma unroll
    for (int r = 0; r < 8; r++) {
      float v = acc[nt][r] + bb;
      v = v / (1.f + __expf(-v));   // silu
      mid[(r + hi * 8) * 64 + col] = (F16)v;
    }
  }
}

// +bias, store ef (f32), attention logits e = sum_d a[h,d]*leaky_relu(f,0.2)
// (head == nt because D==16), plus global softmax upper bound via atomicMax.
__device__ inline void attn_store(v8f* acc, const float* b2, const float* avec,
                                  float* ef, float* ebuf, float* maxv,
                                  int ebase, int wv, int lane, int M) {
  const int hi = lane >> 4, lo = lane & 15;
  float m = -3.0e38f;
#pragma unroll
  for (int nt = 0; nt < 4; nt++) {
    const int col = nt * 16 + lo;
    const float av = avec[col];
    const float bb = b2[col];
    float es[8];
#pragma unroll
    for (int r = 0; r < 8; r++) {
      float v = acc[nt][r] + bb;
      const int e = ebase + wv * 16 + r + hi * 8;
      if (e < M) ef[(size_t)e * 64 + col] = v;
      const float lr = v > 0.f ? v : 0.2f * v;
      es[r] = av * lr;
    }
#pragma unroll
    for (int off = 1; off < 16; off <<= 1)
#pragma unroll
      for (int r = 0; r < 8; r++) es[r] += __shfl_xor(es[r], off);
#pragma unroll
    for (int r = 0; r < 8; r++) {
      const int e = ebase + wv * 16 + r + hi * 8;
      if (lo == 0 && e < M) ebuf[(size_t)e * 4 + nt] = es[r];
      m = fmaxf(m, es[r]);
    }
  }
#pragma unroll
  for (int off = 1; off < 32; off <<= 1) m = fmaxf(m, __shfl_xor(m, off));
  if (lane == 0) atomicMax((unsigned int*)maxv, __float_as_uint(fmaxf(m, 0.f)));
}

// ---------------------------------------------------------------------------
// Edge branch: gather + 161->64 silu -> 64->64 MLP + attention logits.
// 64 edges / block, 128 threads (4 waves x 16 rows), K padded to 192.
// ---------------------------------------------------------------------------
__global__ void __launch_bounds__(128) edge_feat_kernel(
    const float* __restrict__ x, const int* __restrict__ eidx,
    const float* __restrict__ eattr, const float* __restrict__ pos,
    const float* __restrict__ W1, const float* __restrict__ b1,
    const float* __restrict__ W2, const float* __restrict__ b2,
    const float* __restrict__ avec,
    float* __restrict__ ef, float* __restrict__ ebuf, float* __restrict__ diffb,
    float* __restrict__ maxv, int E) {
  __shared__ F16 sfeat[64 * 192];
  __shared__ F16 sw[192 * 64];
  __shared__ F16 smid[64 * 64];
  __shared__ float srad[64];
  __shared__ int sei[64], sej[64];
  const int tid = threadIdx.x;
  const int e0 = blockIdx.x * 64;
  if (tid < 64) {
    const int e = e0 + tid;
    const bool ok = e < E;
    const int i = ok ? eidx[e] : 0;
    const int j = ok ? eidx[(size_t)E + e] : 0;
    sei[tid] = i; sej[tid] = j;
    const float d0 = pos[j * 3 + 0] - pos[i * 3 + 0];
    const float d1 = pos[j * 3 + 1] - pos[i * 3 + 1];
    const float d2 = pos[j * 3 + 2] - pos[i * 3 + 2];
    const float r = sqrtf(d0 * d0 + d1 * d1 + d2 * d2);
    srad[tid] = r;
    if (ok) {
      const float inv = 1.f / (sqrtf(r) + 1e-16f);
      diffb[(size_t)e * 3 + 0] = d0 * inv;
      diffb[(size_t)e * 3 + 1] = d1 * inv;
      diffb[(size_t)e * 3 + 2] = d2 * inv;
    }
  }
  __syncthreads();
  {
    const int r = tid >> 1, hf = tid & 1;
    const int e = e0 + r;
    const bool ok = e < E;
    const float* xi = x + (size_t)sei[r] * 64;
    const float* xj = x + (size_t)sej[r] * 64;
    for (int c = hf * 96; c < hf * 96 + 96; c++) {
      float v = 0.f;
      if (ok) {
        if (c < 64) v = xi[c];
        else if (c < 128) v = xj[c - 64];
        else if (c < 160) v = eattr[(size_t)e * 32 + (c - 128)];
        else if (c == 160) v = srad[r];
      }
      sfeat[r * 192 + c] = (F16)v;
    }
    const int col = tid >> 1;
    for (int k = hf * 96; k < hf * 96 + 96; k++)
      sw[col * 192 + k] = (k < 161) ? (F16)W1[k * 64 + col] : (F16)0.f;
  }
  __syncthreads();
  const int wv = tid >> 5, lane = tid & 31;
  v8f acc[4] = {};
  mma_tile<192>(sfeat + wv * 16 * 192, sw, acc, lane);
  epilogue_silu(acc, b1, smid + wv * 16 * 64, lane);
  __syncthreads();
  {
    const int col = tid >> 1, hf = tid & 1;
    for (int k = hf * 32; k < hf * 32 + 32; k++)
      sw[col * 64 + k] = (F16)W2[k * 64 + col];
  }
  __syncthreads();
  v8f acc2[4] = {};
  mma_tile<64>(smid + wv * 16 * 64, sw, acc2, lane);
  attn_store(acc2, b2, avec, ef, ebuf, maxv, e0, wv, lane, E);
}

// ---------------------------------------------------------------------------
// Angle branch: 32 angles / block, 64 threads (2 waves), K padded to 288.
// ---------------------------------------------------------------------------
__global__ void __launch_bounds__(64) angle_feat_kernel(
    const float* __restrict__ x, const int* __restrict__ aidx,
    const float* __restrict__ aattr, const float* __restrict__ aeattr,
    const float* __restrict__ pos,
    const float* __restrict__ W1, const float* __restrict__ b1,
    const float* __restrict__ W2, const float* __restrict__ b2,
    const float* __restrict__ avec,
    float* __restrict__ ef, float* __restrict__ ebuf, float* __restrict__ diffb,
    float* __restrict__ maxv, int A) {
  __shared__ F16 sfeat[32 * 288];
  __shared__ F16 sw[288 * 64];
  __shared__ F16 smid[32 * 64];
  __shared__ float srad[32];
  __shared__ int sai[32], sah[32], saj[32];
  const int tid = threadIdx.x;
  const int a0 = blockIdx.x * 32;
  if (tid < 32) {
    const int a = a0 + tid;
    const bool ok = a < A;
    const int i = ok ? aidx[a] : 0;
    const int h = ok ? aidx[(size_t)A + a] : 0;
    const int j = ok ? aidx[2 * (size_t)A + a] : 0;
    sai[tid] = i; sah[tid] = h; saj[tid] = j;
    // pos[ah]-pos[ai] + (pos[aj]-pos[ah]) == pos[aj]-pos[ai]
    const float d0 = pos[j * 3 + 0] - pos[i * 3 + 0];
    const float d1 = pos[j * 3 + 1] - pos[i * 3 + 1];
    const float d2 = pos[j * 3 + 2] - pos[i * 3 + 2];
    const float r = sqrtf(d0 * d0 + d1 * d1 + d2 * d2);
    srad[tid] = r;
    if (ok) {
      const float inv = 1.f / (sqrtf(r) + 1e-16f);
      diffb[(size_t)a * 3 + 0] = d0 * inv;
      diffb[(size_t)a * 3 + 1] = d1 * inv;
      diffb[(size_t)a * 3 + 2] = d2 * inv;
    }
  }
  __syncthreads();
  {
    const int r = tid >> 1, hf = tid & 1;
    const int a = a0 + r;
    const bool ok = a < A;
    const float* xi = x + (size_t)sai[r] * 64;
    const float* xh = x + (size_t)sah[r] * 64;
    const float* xj = x + (size_t)saj[r] * 64;
    for (int c = hf * 144; c < hf * 144 + 144; c++) {
      float v = 0.f;
      if (ok) {
        if (c < 64) v = xi[c];
        else if (c < 128) v = xh[c - 64];
        else if (c < 192) v = xj[c - 128];
        else if (c < 256) v = aeattr[(size_t)a * 64 + (c - 192)];
        else if (c < 264) v = aattr[(size_t)a * 8 + (c - 256)] * 0.017453292519943295f;
        else if (c == 264) v = srad[r];
      }
      sfeat[r * 288 + c] = (F16)v;
    }
    const int col = tid;
    for (int k = 0; k < 288; k++)
      sw[(size_t)col * 288 + k] = (k < 265) ? (F16)W1[k * 64 + col] : (F16)0.f;
  }
  __syncthreads();
  const int wv = tid >> 5, lane = tid & 31;
  v8f acc[4] = {};
  mma_tile<288>(sfeat + wv * 16 * 288, sw, acc, lane);
  epilogue_silu(acc, b1, smid + wv * 16 * 64, lane);
  __syncthreads();
  {
    const int col = tid;
    for (int k = 0; k < 64; k++) sw[col * 64 + k] = (F16)W2[k * 64 + col];
  }
  __syncthreads();
  v8f acc2[4] = {};
  mma_tile<64>(smid + wv * 16 * 64, sw, acc2, lane);
  attn_store(acc2, b2, avec, ef, ebuf, maxv, a0, wv, lane, A);
}

// ---------------------------------------------------------------------------
// Softmax normalize: p = exp(e - M'), segment sums via atomics.
// ---------------------------------------------------------------------------
__global__ void attn_norm_kernel(float* __restrict__ ebuf, float* __restrict__ s,
                                 const int* __restrict__ jidx,
                                 const float* __restrict__ maxv, int M) {
  const int t = blockIdx.x * blockDim.x + threadIdx.x;
  if (t < M * 4) {
    const int e = t >> 2, h = t & 3;
    const float mm = maxv[0];
    const float p = __expf(ebuf[t] - mm);
    ebuf[t] = p;
    atomicAdd(&s[(size_t)jidx[e] * 4 + h], p);
  }
}

// ---------------------------------------------------------------------------
// Aggregate: proj = ef * softmax weight -> atomic agg; pos-message MLP (WMMA)
// -> t -> atomic num/cnt. 64 edges / block, 128 threads.
// ---------------------------------------------------------------------------
__global__ void __launch_bounds__(128) aggregate_kernel(
    const float* __restrict__ ef, const float* __restrict__ ebuf,
    const float* __restrict__ s, const int* __restrict__ jidx,
    const float* __restrict__ diffb,
    const float* __restrict__ W1, const float* __restrict__ b1,
    const float* __restrict__ W2,
    float* __restrict__ agg, float* __restrict__ num, float* __restrict__ cnt,
    int M) {
  __shared__ F16 sproj[64 * 64];
  __shared__ F16 sw[64 * 64];
  __shared__ F16 smid[64 * 64];
  __shared__ int sj[64];
  const int tid = threadIdx.x;
  const int e0 = blockIdx.x * 64;
  if (tid < 64) sj[tid] = (e0 + tid < M) ? jidx[e0 + tid] : 0;
  __syncthreads();
  {
    const int r = tid >> 1, hf = tid & 1;
    const int e = e0 + r;
    const bool ok = e < M;
    const int j = sj[r];
    for (int c = hf * 32; c < hf * 32 + 32; c++) {
      float pv = 0.f;
      if (ok) {
        const int hd = c >> 4;
        const float a = ebuf[(size_t)e * 4 + hd] / (s[(size_t)j * 4 + hd] + 1e-16f);
        pv = ef[(size_t)e * 64 + c] * a;
        atomicAdd(&agg[(size_t)j * 64 + c], pv);
      }
      sproj[r * 64 + c] = (F16)pv;
    }
    const int col = tid >> 1;
    for (int k = hf * 32; k < hf * 32 + 32; k++)
      sw[col * 64 + k] = (F16)W1[k * 64 + col];
  }
  __syncthreads();
  const int wv = tid >> 5, lane = tid & 31;
  v8f acc[4] = {};
  mma_tile<64>(sproj + wv * 16 * 64, sw, acc, lane);
  epilogue_silu(acc, b1, smid + wv * 16 * 64, lane);
  __syncthreads();
  if (tid < 64) {
    const int r = tid;
    const int e = e0 + r;
    if (e < M) {
      const int j = sj[r];
      float t = 0.f;
      for (int k = 0; k < 64; k++) t += (float)smid[r * 64 + k] * W2[k];
      atomicAdd(&num[(size_t)j * 3 + 0], diffb[(size_t)e * 3 + 0] * t);
      atomicAdd(&num[(size_t)j * 3 + 1], diffb[(size_t)e * 3 + 1] * t);
      atomicAdd(&num[(size_t)j * 3 + 2], diffb[(size_t)e * 3 + 2] * t);
      atomicAdd(&cnt[j], 1.f);
    }
  }
}

// ---------------------------------------------------------------------------
// Node update MLP: concat(x, agg) [128] -> 64 (silu) -> 64. 64 nodes / block.
// ---------------------------------------------------------------------------
__global__ void __launch_bounds__(128) node_update_kernel(
    const float* __restrict__ x, const float* __restrict__ agg,
    const float* __restrict__ W1, const float* __restrict__ b1,
    const float* __restrict__ W2, const float* __restrict__ b2,
    float* __restrict__ xout, int N) {
  __shared__ F16 sfeat[64 * 128];
  __shared__ F16 sw[128 * 64];
  __shared__ F16 smid[64 * 64];
  const int tid = threadIdx.x;
  const int n0 = blockIdx.x * 64;
  {
    const int r = tid >> 1, hf = tid & 1;
    const int n = n0 + r;
    const bool ok = n < N;
    for (int c = hf * 64; c < hf * 64 + 64; c++) {
      float v = 0.f;
      if (ok) v = (c < 64) ? x[(size_t)n * 64 + c] : agg[(size_t)n * 64 + (c - 64)];
      sfeat[r * 128 + c] = (F16)v;
    }
    const int col = tid >> 1;
    for (int k = hf * 64; k < hf * 64 + 64; k++)
      sw[col * 128 + k] = (F16)W1[k * 64 + col];
  }
  __syncthreads();
  const int wv = tid >> 5, lane = tid & 31, hi = lane >> 4, lo = lane & 15;
  v8f acc[4] = {};
  mma_tile<128>(sfeat + wv * 16 * 128, sw, acc, lane);
  epilogue_silu(acc, b1, smid + wv * 16 * 64, lane);
  __syncthreads();
  {
    const int col = tid >> 1, hf = tid & 1;
    for (int k = hf * 32; k < hf * 32 + 32; k++)
      sw[col * 64 + k] = (F16)W2[k * 64 + col];
  }
  __syncthreads();
  v8f acc2[4] = {};
  mma_tile<64>(smid + wv * 16 * 64, sw, acc2, lane);
#pragma unroll
  for (int nt = 0; nt < 4; nt++) {
    const int col = nt * 16 + lo;
    const float bb = b2[col];
#pragma unroll
    for (int r = 0; r < 8; r++) {
      const int n = n0 + wv * 16 + r + hi * 8;
      if (n < N) xout[(size_t)n * 64 + col] = acc2[nt][r] + bb;
    }
  }
}

// pos + num / max(cnt,1)
__global__ void posup_kernel(const float* __restrict__ pos, const float* __restrict__ num,
                             const float* __restrict__ cnt, float* __restrict__ pout, int N) {
  const int n = blockIdx.x * blockDim.x + threadIdx.x;
  if (n < N) {
    const float c = fmaxf(cnt[n], 1.f);
    for (int k = 0; k < 3; k++)
      pout[(size_t)n * 3 + k] = pos[(size_t)n * 3 + k] + num[(size_t)n * 3 + k] / c;
  }
}

// ---------------------------------------------------------------------------
// Fusion: 3 projections (WMMA), 2-way head softmax, residual, LayerNorm(D=16),
// PReLU. 64 nodes / block, 128 threads.
// ---------------------------------------------------------------------------
__global__ void __launch_bounds__(128) fusion_kernel(
    const float* __restrict__ x, const float* __restrict__ xe, const float* __restrict__ xa,
    const float* __restrict__ seW, const float* __restrict__ seb,
    const float* __restrict__ saW, const float* __restrict__ sab,
    const float* __restrict__ sxW, const float* __restrict__ sxb,
    const float* __restrict__ bias, const float* __restrict__ lng,
    const float* __restrict__ lnb, const float* __restrict__ prelu_p,
    float* __restrict__ out, int N) {
  __shared__ F16 fe[64 * 64], fa[64 * 64], fx[64 * 64];
  __shared__ F16 w1[64 * 64], w2[64 * 64], w3[64 * 64];
  const int tid = threadIdx.x;
  const int n0 = blockIdx.x * 64;
  {
    const int r = tid >> 1, hf = tid & 1;
    const int n = n0 + r;
    const bool ok = n < N;
    for (int c = hf * 32; c < hf * 32 + 32; c++) {
      fe[r * 64 + c] = ok ? (F16)xe[(size_t)n * 64 + c] : (F16)0.f;
      fa[r * 64 + c] = ok ? (F16)xa[(size_t)n * 64 + c] : (F16)0.f;
      fx[r * 64 + c] = ok ? (F16)x[(size_t)n * 64 + c] : (F16)0.f;
    }
    const int col = tid >> 1;
    for (int k = hf * 32; k < hf * 32 + 32; k++) {
      w1[col * 64 + k] = (F16)seW[k * 64 + col];
      w2[col * 64 + k] = (F16)saW[k * 64 + col];
      w3[col * 64 + k] = (F16)sxW[k * 64 + col];
    }
  }
  __syncthreads();
  const int wv = tid >> 5, lane = tid & 31, hi = lane >> 4, lo = lane & 15;
  v8f aE[4] = {}, aA[4] = {}, aX[4] = {};
  mma_tile<64>(fe + wv * 16 * 64, w1, aE, lane);
  mma_tile<64>(fa + wv * 16 * 64, w2, aA, lane);
  mma_tile<64>(fx + wv * 16 * 64, w3, aX, lane);
  const float pr = prelu_p[0];
#pragma unroll
  for (int nt = 0; nt < 4; nt++) {
    const int col = nt * 16 + lo;
    const float be = seb[col], ba = sab[col], bx = sxb[col];
#pragma unroll
    for (int r = 0; r < 8; r++) {
      const float xe2 = aE[nt][r] + be;
      const float xa2 = aA[nt][r] + ba;
      const float xx  = aX[nt][r] + bx;
      float pe = xx * xe2, pa = xx * xa2;
#pragma unroll
      for (int off = 1; off < 16; off <<= 1) {
        pe += __shfl_xor(pe, off);
        pa += __shfl_xor(pa, off);
      }
      pe *= 0.25f; pa *= 0.25f;               // scale = 1/sqrt(D)
      const float mm = fmaxf(pe, pa);
      const float we = __expf(pe - mm), wa = __expf(pa - mm);
      const float w0 = we / (we + wa);
      const int n = n0 + wv * 16 + r + hi * 8;
      const float xorig = x[(size_t)(n < N ? n : 0) * 64 + col];
      float xo = w0 * xe2 + (1.f - w0) * xa2 + xorig + bias[lo];
      float s1 = xo, s2 = xo * xo;
#pragma unroll
      for (int off = 1; off < 16; off <<= 1) {
        s1 += __shfl_xor(s1, off);
        s2 += __shfl_xor(s2, off);
      }
      const float mu = s1 * 0.0625f;
      const float var = s2 * 0.0625f - mu * mu;
      float xn = (xo - mu) * rsqrtf(var + 1e-5f) * lng[lo] + lnb[lo];
      xn = xn > 0.f ? xn : pr * xn;
      if (n < N) out[(size_t)n * 64 + col] = xn;
    }
  }
}

// po = pos_e@epW+epb + pos_a@apW+apb + pos@xpW+xpb  (3x3 mats)
__global__ void posfuse_kernel(const float* __restrict__ pos,
                               const float* __restrict__ pe_, const float* __restrict__ pa_,
                               const float* __restrict__ epW, const float* __restrict__ epb,
                               const float* __restrict__ apW, const float* __restrict__ apb,
                               const float* __restrict__ xpW, const float* __restrict__ xpb,
                               float* __restrict__ out, int N) {
  const int n = blockIdx.x * blockDim.x + threadIdx.x;
  if (n < N) {
    float e[3], a[3], p[3];
    for (int m = 0; m < 3; m++) {
      e[m] = pe_[(size_t)n * 3 + m];
      a[m] = pa_[(size_t)n * 3 + m];
      p[m] = pos[(size_t)n * 3 + m];
    }
    for (int k = 0; k < 3; k++) {
      float v = epb[k] + apb[k] + xpb[k];
      for (int m = 0; m < 3; m++)
        v += e[m] * epW[m * 3 + k] + a[m] * apW[m * 3 + k] + p[m] * xpW[m * 3 + k];
      out[(size_t)n * 3 + k] = v;
    }
  }
}

// ---------------------------------------------------------------------------
extern "C" void kernel_launch(void* const* d_in, const int* in_sizes, int n_in,
                              void* d_out, int out_size, void* d_ws, size_t ws_size,
                              hipStream_t stream) {
  const float* x      = (const float*)d_in[0];
  const int*   eidx   = (const int*)  d_in[1];
  const float* eattr  = (const float*)d_in[2];
  const int*   aidx   = (const int*)  d_in[3];
  const float* aattr  = (const float*)d_in[4];
  const float* aeattr = (const float*)d_in[5];
  const float* pos    = (const float*)d_in[6];
  const float* node_W1 = (const float*)d_in[7];
  const float* node_b1 = (const float*)d_in[8];
  const float* node_W2 = (const float*)d_in[9];
  const float* node_b2 = (const float*)d_in[10];
  const float* posm_W1 = (const float*)d_in[11];
  const float* posm_b1 = (const float*)d_in[12];
  const float* posm_W2 = (const float*)d_in[13];
  const float* edge_W1 = (const float*)d_in[14];
  const float* edge_b1 = (const float*)d_in[15];
  const float* edge_W2 = (const float*)d_in[16];
  const float* edge_b2 = (const float*)d_in[17];
  const float* ang_W1  = (const float*)d_in[18];
  const float* ang_b1  = (const float*)d_in[19];
  const float* ang_W2  = (const float*)d_in[20];
  const float* ang_b2  = (const float*)d_in[21];
  const float* se_W = (const float*)d_in[22];
  const float* se_b = (const float*)d_in[23];
  const float* sa_W = (const float*)d_in[24];
  const float* sa_b = (const float*)d_in[25];
  const float* sx_W = (const float*)d_in[26];
  const float* sx_b = (const float*)d_in[27];
  const float* ep_W = (const float*)d_in[28];
  const float* ep_b = (const float*)d_in[29];
  const float* ap_W = (const float*)d_in[30];
  const float* ap_b = (const float*)d_in[31];
  const float* xp_W = (const float*)d_in[32];
  const float* xp_b = (const float*)d_in[33];
  const float* a_edge  = (const float*)d_in[34];
  const float* a_angle = (const float*)d_in[35];
  const float* bias = (const float*)d_in[36];
  const float* ln_g = (const float*)d_in[37];
  const float* ln_b = (const float*)d_in[38];
  const float* prelu = (const float*)d_in[39];

  const int N = in_sizes[0] / 64;
  const int E = in_sizes[1] / 2;
  const int A = in_sizes[3] / 3;
  const size_t M = (size_t)(E > A ? E : A);

  float* ws = (float*)d_ws;
  size_t o = 0;
  float* ef    = ws + o; o += M * 64;   // shared between branches
  float* ebuf  = ws + o; o += M * 4;
  float* diffb = ws + o; o += M * 3;
  float* zero_base = ws + o;
  float* s_e   = ws + o; o += (size_t)N * 4;
  float* agg_e = ws + o; o += (size_t)N * 64;
  float* num_e = ws + o; o += (size_t)N * 3;
  float* cnt_e = ws + o; o += (size_t)N;
  float* s_a   = ws + o; o += (size_t)N * 4;
  float* agg_a = ws + o; o += (size_t)N * 64;
  float* num_a = ws + o; o += (size_t)N * 3;
  float* cnt_a = ws + o; o += (size_t)N;
  float* maxv  = ws + o; o += 2;
  const size_t zero_count = (size_t)(ws + o - zero_base);
  float* xe_  = ws + o; o += (size_t)N * 64;
  float* xa_  = ws + o; o += (size_t)N * 64;
  float* pe_  = ws + o; o += (size_t)N * 3;
  float* pa_  = ws + o; o += (size_t)N * 3;

  float* out = (float*)d_out;

  hipMemsetAsync(zero_base, 0, zero_count * sizeof(float), stream);

  // ---- edge branch ----
  edge_feat_kernel<<<(E + 63) / 64, 128, 0, stream>>>(
      x, eidx, eattr, pos, edge_W1, edge_b1, edge_W2, edge_b2, a_edge,
      ef, ebuf, diffb, maxv + 0, E);
  attn_norm_kernel<<<(E * 4 + 255) / 256, 256, 0, stream>>>(
      ebuf, s_e, eidx + E, maxv + 0, E);
  aggregate_kernel<<<(E + 63) / 64, 128, 0, stream>>>(
      ef, ebuf, s_e, eidx + E, diffb, posm_W1, posm_b1, posm_W2,
      agg_e, num_e, cnt_e, E);
  node_update_kernel<<<(N + 63) / 64, 128, 0, stream>>>(
      x, agg_e, node_W1, node_b1, node_W2, node_b2, xe_, N);
  posup_kernel<<<(N + 255) / 256, 256, 0, stream>>>(pos, num_e, cnt_e, pe_, N);

  // ---- angle branch (reuses ef / ebuf / diffb) ----
  angle_feat_kernel<<<(A + 31) / 32, 64, 0, stream>>>(
      x, aidx, aattr, aeattr, pos, ang_W1, ang_b1, ang_W2, ang_b2, a_angle,
      ef, ebuf, diffb, maxv + 1, A);
  attn_norm_kernel<<<(A * 4 + 255) / 256, 256, 0, stream>>>(
      ebuf, s_a, aidx + 2 * (size_t)A, maxv + 1, A);
  aggregate_kernel<<<(A + 63) / 64, 128, 0, stream>>>(
      ef, ebuf, s_a, aidx + 2 * (size_t)A, diffb, posm_W1, posm_b1, posm_W2,
      agg_a, num_a, cnt_a, A);
  node_update_kernel<<<(N + 63) / 64, 128, 0, stream>>>(
      x, agg_a, node_W1, node_b1, node_W2, node_b2, xa_, N);
  posup_kernel<<<(N + 255) / 256, 256, 0, stream>>>(pos, num_a, cnt_a, pa_, N);

  // ---- fusion ----
  fusion_kernel<<<(N + 63) / 64, 128, 0, stream>>>(
      x, xe_, xa_, se_W, se_b, sa_W, sa_b, sx_W, sx_b,
      bias, ln_g, ln_b, prelu, out, N);
  posfuse_kernel<<<(N + 255) / 256, 256, 0, stream>>>(
      pos, pe_, pa_, ep_W, ep_b, ap_W, ap_b, xp_W, xp_b,
      out + (size_t)N * 64, N);
}
